// AttentiveFPPredictor_76020921140172
// MI455X (gfx1250) — matrix operand order repaired and checked by reading, not compiled, not run on hardware
//
#include <hip/hip_runtime.h>
#include <hip/hip_bf16.h>

// ---------------------------------------------------------------------------
// AttentiveFP forward for MI455X (gfx1250, wave32).
// Dense GEMMs -> v_wmma_f32_16x16x32_bf16 (fp32 accum, bf16 operands).
// Graph ops  -> gathers + wave reductions + f32/u32 global atomics.
// ---------------------------------------------------------------------------

#define NN 250000
#define EE 500000
#define BB 10000
#define HH 128

typedef __bf16 bf16;
typedef bf16  v16bf __attribute__((ext_vector_type(16)));
typedef float v8f   __attribute__((ext_vector_type(8)));
typedef unsigned short u16;
typedef unsigned int   u32;

// ---- helpers ---------------------------------------------------------------

__device__ __forceinline__ u16 f2bf(float f) {
  u32 u = __float_as_uint(f);
  u32 r = 0x7FFFu + ((u >> 16) & 1u);     // round-to-nearest-even
  return (u16)((u + r) >> 16);
}

__device__ __forceinline__ float wred(float s) {
#pragma unroll
  for (int o = 16; o > 0; o >>= 1) s += __shfl_xor(s, o, 32);
  return s;
}

// order-preserving float->uint key for atomicMax on floats
__device__ __forceinline__ u32 fkey(float f) {
  u32 u = __float_as_uint(f);
  return (u & 0x80000000u) ? ~u : (u | 0x80000000u);
}
__device__ __forceinline__ float funkey(u32 k) {
  u32 u = (k & 0x80000000u) ? (k & 0x7FFFFFFFu) : ~k;
  return __uint_as_float(u);
}

__device__ __forceinline__ float lrelu01(float v) { return v > 0.f ? v : 0.01f * v; }

// ---- weight fp32 -> bf16 (K padded to kp with zeros) ----------------------

__global__ void k_w2bf(u16* __restrict__ dst, const float* __restrict__ src,
                       int rows, int kc, int ld, int off, int kp) {
  int i = blockIdx.x * blockDim.x + threadIdx.x;
  if (i >= rows * kp) return;
  int r = i / kp, k = i % kp;
  dst[i] = (k < kc) ? f2bf(src[(size_t)r * ld + off + k]) : (u16)0;
}

// ---- WMMA GEMM: C[M,Nc] = act(A[M,K] @ W^T + bias) ------------------------
// W is bf16 [Nc, kp] row-major (K padded). One wave per 16-row tile, NT
// 16-col tiles per wave (A fragment reused NT times per K step).
// act: 0=none, 1=leaky_relu(0.01), 2=relu.

template <int NT>
__global__ __launch_bounds__(32) void k_gemm(const float* __restrict__ A, int lda,
                                             const u16* __restrict__ W, int kp,
                                             const float* __restrict__ bias,
                                             float* __restrict__ C, int ldc,
                                             int K, int act) {
  const int mt = blockIdx.x;
  const int nt0 = blockIdx.y * NT;
  const int lane = threadIdx.x;
  const int half = lane >> 4;     // 0 | 1
  const int l15  = lane & 15;

  v8f acc[NT];
#pragma unroll
  for (int t = 0; t < NT; ++t)
#pragma unroll
    for (int r = 0; r < 8; ++r) acc[t][r] = 0.f;

  const float* arow = A + (size_t)(mt * 16 + l15) * lda;

  for (int k0 = 0; k0 < K; k0 += 32) {
    // A fragment (16-bit A 16x32 layout, ISA 7.12.2):
    // elems 0..7  -> K = k0 + half*8 + e
    // elems 8..15 -> K = k0 + 16 + half*8 + e
    union { v16bf v; u16 u[16]; } af;
    const int kb = k0 + half * 8;
#pragma unroll
    for (int e = 0; e < 8; ++e) { int k = kb + e;      af.u[e]     = (k < K) ? f2bf(arow[k]) : (u16)0; }
#pragma unroll
    for (int e = 0; e < 8; ++e) { int k = kb + 16 + e; af.u[8 + e] = (k < K) ? f2bf(arow[k]) : (u16)0; }

#pragma unroll
    for (int t = 0; t < NT; ++t) {
      // B fragment: same K order as A (elems 0..7 -> k0+half*8+e, 8..15 -> k0+16+half*8+e)
      union { v16bf v; u16 u[16]; } bfv;
      const u16* wr = W + (size_t)((nt0 + t) * 16 + l15) * kp + k0 + half * 8;
#pragma unroll
      for (int e = 0; e < 16; ++e) bfv.u[e] = wr[(e & 8) * 2 + (e & 7)];
      acc[t] = __builtin_amdgcn_wmma_f32_16x16x32_bf16(
          false, af.v, false, bfv.v, (short)0, acc[t], false, false);
    }
  }

#pragma unroll
  for (int t = 0; t < NT; ++t) {
    int ncol = (nt0 + t) * 16 + l15;
    float b = bias ? bias[ncol] : 0.f;
#pragma unroll
    for (int r = 0; r < 8; ++r) {
      int row = mt * 16 + half * 8 + r;   // C layout: VGPR r -> M = r (+8 for upper lanes)
      float v = acc[t][r] + b;
      if (act == 1) v = lrelu01(v);
      else if (act == 2) v = v > 0.f ? v : 0.f;
      C[(size_t)row * ldc + ncol] = v;
    }
  }
}

// ---- elementwise / graph kernels ------------------------------------------

__global__ void k_set(float* p, long n, float v) {
  long i = (long)blockIdx.x * blockDim.x + threadIdx.x;
  if (i < n) p[i] = v;
}

__global__ void k_bias_rows(float* __restrict__ p, const float* __restrict__ bias, long rows) {
  long i = (long)blockIdx.x * blockDim.x + threadIdx.x;
  if (i < rows * HH) p[i] = bias[i & (HH - 1)];
}

// act: 0 = relu, 1 = elu
__global__ void k_act(float* p, long n, int mode) {
  long i = (long)blockIdx.x * blockDim.x + threadIdx.x;
  if (i >= n) return;
  float v = p[i];
  p[i] = (mode == 0) ? (v > 0.f ? v : 0.f) : (v > 0.f ? v : expm1f(v));
}

// per-row dot: out[r] = X[r,:] . w   (one wave per row)
__global__ __launch_bounds__(256) void k_rowdot(const float* __restrict__ X,
                                                const float* __restrict__ w,
                                                float* __restrict__ out, int rows) {
  int r = blockIdx.x * 8 + (threadIdx.x >> 5);
  int lane = threadIdx.x & 31;
  if (r >= rows) return;
  const float* xr = X + (size_t)r * HH;
  float s = 0.f;
  for (int h = lane; h < HH; h += 32) s += xr[h] * w[h];
  s = wred(s);
  if (lane == 0) out[r] = s;
}

// GATEConv attention logits: one wave per edge
__global__ __launch_bounds__(256) void k_gate_logits(
    const float* __restrict__ U, const float* __restrict__ V,
    const float* __restrict__ attl, const float* __restrict__ R,
    const int* __restrict__ src, const int* __restrict__ dst,
    float* __restrict__ L, u32* __restrict__ MK, int E) {
  int e = blockIdx.x * 8 + (threadIdx.x >> 5);
  int lane = threadIdx.x & 31;
  if (e >= E) return;
  const float* u = U + (size_t)src[e] * HH;
  const float* v = V + (size_t)e * HH;
  float acc = 0.f;
  for (int h = lane; h < HH; h += 32) acc += lrelu01(u[h] + v[h]) * attl[h];
  acc = wred(acc);
  if (lane == 0) {
    int d = dst[e];
    float lg = lrelu01(acc + R[d]);
    L[e] = lg;
    atomicMax(&MK[d], fkey(lg));
  }
}

// generic scalar-pair logits: lg = lrelu(a[ia ? ia[i] : i] + b[ib[i]])
__global__ void k_pair_logits(const float* __restrict__ a, const int* __restrict__ ia,
                              const float* __restrict__ b, const int* __restrict__ ib,
                              float* __restrict__ L, u32* __restrict__ MK, int n) {
  int i = blockIdx.x * blockDim.x + threadIdx.x;
  if (i >= n) return;
  int s = ia ? ia[i] : i;
  int d = ib[i];
  float lg = lrelu01(a[s] + b[d]);
  L[i] = lg;
  atomicMax(&MK[d], fkey(lg));
}

// L[i] = exp(L[i] - max[seg]); S[seg] += L[i]
__global__ void k_expnorm(float* __restrict__ L, const u32* __restrict__ MK,
                          float* __restrict__ S, const int* __restrict__ seg, int n) {
  int i = blockIdx.x * blockDim.x + threadIdx.x;
  if (i >= n) return;
  int d = seg[i];
  float ev = expf(L[i] - funkey(MK[d]));
  L[i] = ev;
  atomicAdd(&S[d], ev);
}

// T[dst,:] += Y[src,:] * (L[i] / (S[dst]+1e-16)); 128 threads per item
__global__ __launch_bounds__(256) void k_scatter(const float* __restrict__ Y,
                                                 const float* __restrict__ L,
                                                 const float* __restrict__ S,
                                                 const int* __restrict__ idx_src,
                                                 const int* __restrict__ idx_dst,
                                                 float* __restrict__ T, int items) {
  int it = blockIdx.x * 2 + (threadIdx.x >> 7);
  int ch = threadIdx.x & 127;
  if (it >= items) return;
  int s = idx_src ? idx_src[it] : it;
  int d = idx_dst[it];
  float alpha = L[it] / (S[d] + 1e-16f);
  atomicAdd(&T[(size_t)d * HH + ch], Y[(size_t)s * HH + ch] * alpha);
}

// OUT[batch[i],:] += X[i,:]
__global__ void k_segsum(const float* __restrict__ X, const int* __restrict__ batch,
                         float* __restrict__ OUT, long n) {
  long i = (long)blockIdx.x * blockDim.x + threadIdx.x;
  if (i >= n * HH) return;
  long row = i >> 7;
  int ch = (int)(i & (HH - 1));
  atomicAdd(&OUT[(size_t)batch[row] * HH + ch], X[i]);
}

// torch GRUCell combine + relu, in-place on hidden X.
// G1 = xg@Wih^T + bih (gate input from new message), G2 = X@Whh^T + bhh.
__global__ void k_gru(const float* __restrict__ G1, const float* __restrict__ G2,
                      float* __restrict__ X, long rows) {
  long i = (long)blockIdx.x * blockDim.x + threadIdx.x;
  if (i >= rows * HH) return;
  long row = i >> 7;
  int c = (int)(i & (HH - 1));
  size_t b3 = (size_t)row * 3 * HH;
  float ir = G1[b3 + c],           hr = G2[b3 + c];
  float iz = G1[b3 + HH + c],      hz = G2[b3 + HH + c];
  float in_ = G1[b3 + 2 * HH + c], hn = G2[b3 + 2 * HH + c];
  float r = 1.f / (1.f + expf(-(ir + hr)));
  float z = 1.f / (1.f + expf(-(iz + hz)));
  float n = tanhf(in_ + r * hn);
  float hp = X[i];
  float o = (1.f - z) * n + z * hp;
  X[i] = o > 0.f ? o : 0.f;
}

// head: d_out[b] = OUT2[b,:].Whead + bhead  (one wave per row)
__global__ __launch_bounds__(256) void k_head(const float* __restrict__ X,
                                              const float* __restrict__ w,
                                              const float* __restrict__ bh,
                                              float* __restrict__ out, int rows) {
  int r = blockIdx.x * 8 + (threadIdx.x >> 5);
  int lane = threadIdx.x & 31;
  if (r >= rows) return;
  const float* xr = X + (size_t)r * HH;
  float s = 0.f;
  for (int h = lane; h < HH; h += 32) s += xr[h] * w[h];
  s = wred(s);
  if (lane == 0) out[r] = s + bh[0];
}

// ---- workspace layout (float units) ---------------------------------------
static constexpr size_t F_X   = 0;                          // N*H   persistent atom state
static constexpr size_t F_A   = F_X  + (size_t)NN * HH;     // N*H   u / w2x / xl / xs
static constexpr size_t F_E   = F_A  + (size_t)NN * HH;     // E*H   v ; first N*H reused as H accum
static constexpr size_t F_G1  = F_E  + (size_t)EE * HH;     // N*3H  gi
static constexpr size_t F_G2  = F_G1 + (size_t)NN * 3 * HH; // N*3H  gh
static constexpr size_t F_R   = F_G2 + (size_t)NN * 3 * HH; // N
static constexpr size_t F_MK  = F_R  + NN;                  // N (u32 keys)
static constexpr size_t F_S   = F_MK + NN;                  // N
static constexpr size_t F_L   = F_S  + NN;                  // E (also atom logits)
static constexpr size_t F_SS  = F_L  + EE;                  // N
static constexpr size_t F_SD  = F_SS + NN;                  // N
static constexpr size_t F_SA  = F_SD + NN;                  // N
static constexpr size_t F_TB  = F_SA + NN;                  // B
static constexpr size_t F_OUT = F_TB + BB;                  // B*H
static constexpr size_t F_OD  = F_OUT + (size_t)BB * HH;    // B*H
static constexpr size_t F_HM  = F_OD  + (size_t)BB * HH;    // B*H
static constexpr size_t F_GM1 = F_HM  + (size_t)BB * HH;    // B*3H
static constexpr size_t F_GM2 = F_GM1 + (size_t)BB * 3 * HH;// B*3H
static constexpr size_t F_O2  = F_GM2 + (size_t)BB * 3 * HH;// B*H
static constexpr size_t F_WB  = F_O2  + (size_t)BB * HH;    // bf16 weight pool (u16)

// bf16 weight pool sub-offsets (u16 units)
static constexpr size_t W_LIN1 = 0;                   // 128 x 64  (K=39)
static constexpr size_t W_W1X  = W_LIN1 + 128 * 64;   // 128 x 128
static constexpr size_t W_W1E  = W_W1X  + 128 * 128;  // 128 x 32  (K=10)
static constexpr size_t W_W2   = W_W1E  + 128 * 32;   // 128 x 128
static constexpr size_t W_G1I  = W_W2   + 128 * 128;  // 384 x 128
static constexpr size_t W_G1H  = W_G1I  + 384 * 128;
static constexpr size_t W_CONV = W_G1H  + 384 * 128;
static constexpr size_t W_G2I  = W_CONV + 128 * 128;
static constexpr size_t W_G2H  = W_G2I  + 384 * 128;
static constexpr size_t W_MOL  = W_G2H  + 384 * 128;
static constexpr size_t W_GMI  = W_MOL  + 128 * 128;
static constexpr size_t W_GMH  = W_GMI  + 384 * 128;
static constexpr size_t W_LIN2 = W_GMH  + 384 * 128;

static inline int cdiv(long a, int b) { return (int)((a + b - 1) / b); }

// C has 128 cols
static void gemm128(hipStream_t st, const float* A, int lda, const u16* W, int kp,
                    const float* bias, float* C, int M, int K, int act) {
  k_gemm<8><<<dim3(M / 16, 1), 32, 0, st>>>(A, lda, W, kp, bias, C, 128, K, act);
}

extern "C" void kernel_launch(void* const* d_in, const int* in_sizes, int n_in,
                              void* d_out, int out_size, void* d_ws, size_t ws_size,
                              hipStream_t stream) {
  (void)in_sizes; (void)n_in; (void)out_size; (void)ws_size;
  const float* x_in   = (const float*)d_in[0];
  const float* eattr  = (const float*)d_in[1];
  const float* Wlin1  = (const float*)d_in[2];
  const float* blin1  = (const float*)d_in[3];
  const float* gW1    = (const float*)d_in[4];
  const float* gW2    = (const float*)d_in[5];
  const float* g_al   = (const float*)d_in[6];
  const float* g_ar   = (const float*)d_in[7];
  const float* g_b    = (const float*)d_in[8];
  const float* g1Wih  = (const float*)d_in[9];
  const float* g1Whh  = (const float*)d_in[10];
  const float* g1bih  = (const float*)d_in[11];
  const float* g1bhh  = (const float*)d_in[12];
  const float* cW     = (const float*)d_in[13];
  const float* c_as   = (const float*)d_in[14];
  const float* c_ad   = (const float*)d_in[15];
  const float* c_b    = (const float*)d_in[16];
  const float* g2Wih  = (const float*)d_in[17];
  const float* g2Whh  = (const float*)d_in[18];
  const float* g2bih  = (const float*)d_in[19];
  const float* g2bhh  = (const float*)d_in[20];
  const float* mW     = (const float*)d_in[21];
  const float* m_as   = (const float*)d_in[22];
  const float* m_ad   = (const float*)d_in[23];
  const float* m_b    = (const float*)d_in[24];
  const float* gmWih  = (const float*)d_in[25];
  const float* gmWhh  = (const float*)d_in[26];
  const float* gmbih  = (const float*)d_in[27];
  const float* gmbhh  = (const float*)d_in[28];
  const float* Wlin2  = (const float*)d_in[29];
  const float* blin2  = (const float*)d_in[30];
  const float* Whead  = (const float*)d_in[31];
  const float* bhead  = (const float*)d_in[32];
  const int*   eidx   = (const int*)d_in[33];
  const int*   batch  = (const int*)d_in[34];
  const int* src = eidx;        // (2,E) row-major: row 0 = src(j), row 1 = dst(i)
  const int* dst = eidx + EE;

  float* ws = (float*)d_ws;
  float* X   = ws + F_X;
  float* Abf = ws + F_A;         // u / w2x / xl / xs
  float* V   = ws + F_E;         // per-edge gate features
  float* Hac = ws + F_E;         // message accumulator (reuses V space)
  float* G1  = ws + F_G1;
  float* G2  = ws + F_G2;
  float* R   = ws + F_R;
  u32*   MK  = (u32*)(ws + F_MK);
  float* S   = ws + F_S;
  float* L   = ws + F_L;
  float* SSc = ws + F_SS;
  float* SDc = ws + F_SD;
  float* SA  = ws + F_SA;
  float* TB  = ws + F_TB;
  float* OUT = ws + F_OUT;
  float* OD  = ws + F_OD;
  float* HM  = ws + F_HM;
  float* GM1 = ws + F_GM1;
  float* GM2 = ws + F_GM2;
  float* O2  = ws + F_O2;
  u16*   WB  = (u16*)(ws + F_WB);

  // ---- convert all weights to padded bf16 once ----
  auto cvt = [&](size_t off, const float* sw, int rows, int kc, int ld, int coff, int kp) {
    k_w2bf<<<cdiv((long)rows * kp, 256), 256, 0, stream>>>(WB + off, sw, rows, kc, ld, coff, kp);
  };
  cvt(W_LIN1, Wlin1, 128, 39, 39, 0, 64);
  cvt(W_W1X, gW1, 128, 128, 138, 0, 128);
  cvt(W_W1E, gW1, 128, 10, 138, 128, 32);
  cvt(W_W2,  gW2, 128, 128, 128, 0, 128);
  cvt(W_G1I, g1Wih, 384, 128, 128, 0, 128);
  cvt(W_G1H, g1Whh, 384, 128, 128, 0, 128);
  cvt(W_CONV, cW,   128, 128, 128, 0, 128);
  cvt(W_G2I, g2Wih, 384, 128, 128, 0, 128);
  cvt(W_G2H, g2Whh, 384, 128, 128, 0, 128);
  cvt(W_MOL, mW,    128, 128, 128, 0, 128);
  cvt(W_GMI, gmWih, 384, 128, 128, 0, 128);
  cvt(W_GMH, gmWhh, 384, 128, 128, 0, 128);
  cvt(W_LIN2, Wlin2, 128, 128, 128, 0, 128);

  // ---- lin1: X = lrelu(x @ Wlin1^T + b) ----
  gemm128(stream, x_in, 39, WB + W_LIN1, 64, blin1, X, NN, 39, 1);

  // ---- GATEConv ----
  gemm128(stream, X, 128, WB + W_W1X, 128, nullptr, Abf, NN, 128, 0);   // U = X@W1x^T
  gemm128(stream, eattr, 10, WB + W_W1E, 32, nullptr, V, EE, 10, 0);    // V = e@W1e^T
  k_rowdot<<<cdiv(NN, 8), 256, 0, stream>>>(X, g_ar, R, NN);            // R = X.att_r
  k_set<<<cdiv(NN, 256), 256, 0, stream>>>((float*)MK, NN, 0.f);        // key 0 == -inf
  k_set<<<cdiv(NN, 256), 256, 0, stream>>>(S, NN, 0.f);
  k_gate_logits<<<cdiv(EE, 8), 256, 0, stream>>>(Abf, V, g_al, R, src, dst, L, MK, EE);
  k_expnorm<<<cdiv(EE, 256), 256, 0, stream>>>(L, MK, S, dst, EE);
  gemm128(stream, X, 128, WB + W_W2, 128, nullptr, Abf, NN, 128, 0);    // W2X (overwrites U)
  k_bias_rows<<<cdiv((long)NN * HH, 256), 256, 0, stream>>>(Hac, g_b, NN);
  k_scatter<<<EE / 2, 256, 0, stream>>>(Abf, L, S, src, dst, Hac, EE);
  k_act<<<cdiv((long)NN * HH, 256), 256, 0, stream>>>(Hac, (long)NN * HH, 1);  // elu
  k_gemm<12><<<dim3(NN / 16, 2), 32, 0, stream>>>(Hac, 128, WB + W_G1I, 128, g1bih, G1, 384, 128, 0);
  k_gemm<12><<<dim3(NN / 16, 2), 32, 0, stream>>>(X,   128, WB + W_G1H, 128, g1bhh, G2, 384, 128, 0);
  k_gru<<<cdiv((long)NN * HH, 256), 256, 0, stream>>>(G1, G2, X, NN);

  // ---- atom GATConv ----
  gemm128(stream, X, 128, WB + W_CONV, 128, nullptr, Abf, NN, 128, 0);  // XL
  k_rowdot<<<cdiv(NN, 8), 256, 0, stream>>>(Abf, c_as, SSc, NN);
  k_rowdot<<<cdiv(NN, 8), 256, 0, stream>>>(Abf, c_ad, SDc, NN);
  k_set<<<cdiv(NN, 256), 256, 0, stream>>>((float*)MK, NN, 0.f);
  k_set<<<cdiv(NN, 256), 256, 0, stream>>>(S, NN, 0.f);
  k_pair_logits<<<cdiv(EE, 256), 256, 0, stream>>>(SSc, src, SDc, dst, L, MK, EE);
  k_expnorm<<<cdiv(EE, 256), 256, 0, stream>>>(L, MK, S, dst, EE);
  k_bias_rows<<<cdiv((long)NN * HH, 256), 256, 0, stream>>>(Hac, c_b, NN);
  k_scatter<<<EE / 2, 256, 0, stream>>>(Abf, L, S, src, dst, Hac, EE);
  k_act<<<cdiv((long)NN * HH, 256), 256, 0, stream>>>(Hac, (long)NN * HH, 0);  // relu
  k_gemm<12><<<dim3(NN / 16, 2), 32, 0, stream>>>(Hac, 128, WB + W_G2I, 128, g2bih, G1, 384, 128, 0);
  k_gemm<12><<<dim3(NN / 16, 2), 32, 0, stream>>>(X,   128, WB + W_G2H, 128, g2bhh, G2, 384, 128, 0);
  k_gru<<<cdiv((long)NN * HH, 256), 256, 0, stream>>>(G1, G2, X, NN);

  // ---- molecule readout ----
  k_set<<<cdiv((long)BB * HH, 256), 256, 0, stream>>>(OUT, (long)BB * HH, 0.f);
  k_segsum<<<cdiv((long)NN * HH, 256), 256, 0, stream>>>(X, batch, OUT, NN);
  k_act<<<cdiv((long)BB * HH, 256), 256, 0, stream>>>(OUT, (long)BB * HH, 0);  // relu
  gemm128(stream, X, 128, WB + W_MOL, 128, nullptr, Abf, NN, 128, 0);   // XS (x is loop-invariant)
  k_rowdot<<<cdiv(NN, 8), 256, 0, stream>>>(Abf, m_as, SA, NN);

  for (int t = 0; t < 2; ++t) {
    gemm128(stream, OUT, 128, WB + W_MOL, 128, nullptr, OD, BB, 128, 0);
    k_rowdot<<<cdiv(BB, 8), 256, 0, stream>>>(OD, m_ad, TB, BB);
    k_set<<<cdiv(BB, 256), 256, 0, stream>>>((float*)MK, BB, 0.f);
    k_set<<<cdiv(BB, 256), 256, 0, stream>>>(S, BB, 0.f);
    k_pair_logits<<<cdiv(NN, 256), 256, 0, stream>>>(SA, nullptr, TB, batch, L, MK, NN);
    k_expnorm<<<cdiv(NN, 256), 256, 0, stream>>>(L, MK, S, batch, NN);
    k_bias_rows<<<cdiv((long)BB * HH, 256), 256, 0, stream>>>(HM, m_b, BB);
    k_scatter<<<NN / 2, 256, 0, stream>>>(Abf, L, S, nullptr, batch, HM, NN);
    k_act<<<cdiv((long)BB * HH, 256), 256, 0, stream>>>(HM, (long)BB * HH, 1);  // elu
    k_gemm<12><<<dim3(BB / 16, 2), 32, 0, stream>>>(HM,  128, WB + W_GMI, 128, gmbih, GM1, 384, 128, 0);
    k_gemm<12><<<dim3(BB / 16, 2), 32, 0, stream>>>(OUT, 128, WB + W_GMH, 128, gmbhh, GM2, 384, 128, 0);
    k_gru<<<cdiv((long)BB * HH, 256), 256, 0, stream>>>(GM1, GM2, OUT, BB);
  }

  // ---- lin2 + head ----
  gemm128(stream, OUT, 128, WB + W_LIN2, 128, blin2, O2, BB, 128, 0);
  k_head<<<cdiv(BB, 8), 256, 0, stream>>>(O2, Whead, bhead, (float*)d_out, BB);
}
